// IFM_38036230374098
// MI455X (gfx1250) — matrix operand order repaired and checked by reading, not compiled, not run on hardware
//
#include <hip/hip_runtime.h>
#include <math.h>

// ---------------- constants ----------------
#define BATCH 4
#define CHN   256
#define NPIX  16384          // 128*128
#define NHEAD 8
#define HDIM  32
#define NWIN  1024           // BATCH * 16 * 16
#define NTOK  64

typedef __bf16 bf16;
typedef unsigned int u32;
typedef __attribute__((ext_vector_type(16))) __bf16 v16bf;
typedef __attribute__((ext_vector_type(8)))  __bf16 v8bf;
typedef __attribute__((ext_vector_type(8)))  float  v8f;
typedef __attribute__((ext_vector_type(4)))  u32    v4u;
typedef __attribute__((ext_vector_type(8)))  int    v8i;
typedef __attribute__((ext_vector_type(4)))  int    v4i;

// load 16 bf16 as two 16-byte chunks (A-fragment helper per ISA 7.12.2)
__device__ __forceinline__ v16bf load16_2x8(const bf16* p0, const bf16* p1) {
    v8bf lo = *(const v8bf*)p0;
    v8bf hi = *(const v8bf*)p1;
    v16bf r;
#pragma unroll
    for (int i = 0; i < 8; ++i) { r[i] = lo[i]; r[8 + i] = hi[i]; }
    return r;
}

__device__ __forceinline__ float sigmoidf_(float x) { return 1.0f / (1.0f + __expf(-x)); }

// LDS byte offset of a __shared__ object (generic -> AS(3) -> int)
__device__ __forceinline__ u32 lds_offset_of(const void* p) {
    return (u32)(unsigned long long)(const __attribute__((address_space(3))) char*)p;
}

// Tensor Data Mover: async 2-D tile load (bf16 elements) into LDS.
// D# per CDNA5 ISA: group0 = {count|flags, lds_addr, global_addr lo/hi|type=2},
// group1 = {data_size, tensor dims, tile dims, dim0 stride}.
// amdgpu-toolchain (clang-23) 6-arg builtin form.
__device__ __forceinline__ void tdm_load_2d_bf16(u32 lds_off, const bf16* gptr,
                                                 u32 tensor_d0, u32 tensor_d1,
                                                 u32 tile_d0, u32 tile_d1, u32 stride0) {
    unsigned long long ga = (unsigned long long)(size_t)gptr;
    v4u g0;
    g0[0] = 1u;                                             // count=1, user mode
    g0[1] = lds_off;                                        // lds_addr (bytes)
    g0[2] = (u32)(ga & 0xffffffffu);                        // global_addr[31:0]
    g0[3] = (u32)((ga >> 32) & 0x1ffffffu) | (2u << 30);    // global_addr[56:32] | type=2
    v8i g1;
    g1[0] = (int)(1u << 16);                                // data_size=1 (2 bytes)
    g1[1] = (int)((tensor_d0 & 0xffffu) << 16);             // tensor_dim0[15:0] @ bits63:48
    g1[2] = (int)((tensor_d0 >> 16) | ((tensor_d1 & 0xffffu) << 16));
    g1[3] = (int)((tensor_d1 >> 16) | (tile_d0 << 16));     // tile_dim0 @ bits127:112
    g1[4] = (int)(tile_d1 & 0xffffu);                       // tile_dim1 (tile_dim2=0)
    g1[5] = (int)stride0;                                   // tensor_dim0_stride lo32
    g1[6] = 0;
    g1[7] = 0;
    v4i gz4 = {0, 0, 0, 0};
    v8i gz8 = {0, 0, 0, 0, 0, 0, 0, 0};
    __builtin_amdgcn_tensor_load_to_lds(g0, g1, gz4, gz4, gz8, 0);
}

// ---------------- f32 -> bf16 convert ----------------
__global__ void k_f2b(const float* __restrict__ in, bf16* __restrict__ out, long long n) {
    long long i = (long long)blockIdx.x * blockDim.x + threadIdx.x;
    if (i < n) out[i] = (bf16)in[i];
}

// f32 NCHW -> bf16 NHWC transpose ([b][C][NPIX] -> [b][NPIX][C])
__global__ void k_f2bT(const float* __restrict__ in, bf16* __restrict__ out, int C) {
    long long idx = (long long)blockIdx.x * blockDim.x + threadIdx.x;
    if (idx >= (long long)BATCH * C * NPIX) return;
    int pix = (int)(idx % NPIX);
    int ch  = (int)((idx / NPIX) % C);
    int b   = (int)(idx / ((long long)C * NPIX));
    out[((size_t)b * NPIX + pix) * C + ch] = (bf16)in[idx];
}

// ---------------- bf16 WMMA GEMM with TDM-staged B tiles ----------------
// C[bz][M][N] = A[M][K] * Bt[bz][N][K]^T   (A row-major weights, Bt = activations NHWC)
// block = 256 threads (8 waves) -> 32x256 output tile; wave -> 16x64 (4 wmma tiles)
// B tile (256 N x 32 K bf16, 16KB) DMA'd into LDS by the TDM, double buffered.
template<bool OUTF32>
__global__ __launch_bounds__(256) void k_gemm(const bf16* __restrict__ A,
                                              const bf16* __restrict__ Bt,
                                              float* __restrict__ Cf,
                                              bf16*  __restrict__ Cb,
                                              int M, int N, int K) {
    __shared__ __align__(32) bf16 bufs[2][256 * 32];
    int lane = threadIdx.x & 31, wid = threadIdx.x >> 5;
    int hb = lane >> 4, lr = lane & 15;
    int m0 = blockIdx.y * 32 + (wid & 1) * 16;       // wave's 16-row slab
    int nl0 = (wid >> 1) * 64;                       // wave's 64-col slab (block-local)
    int nblk = blockIdx.x * 256;
    int bz = blockIdx.z;
    const bf16* Bb = Bt + (size_t)bz * N * K;        // [N][K]

    // prologue: TDM load of first B tile
    if (wid == 0) {
        tdm_load_2d_bf16(lds_offset_of(&bufs[0][0]),
                         Bb + (size_t)nblk * K, (u32)K, (u32)N, 32u, 256u, (u32)K);
        __builtin_amdgcn_s_wait_tensorcnt(0);
    }
    __syncthreads();

    v8f acc[4];
    v8f zero = {};
#pragma unroll
    for (int t = 0; t < 4; ++t) acc[t] = zero;

    int nsteps = K >> 5;
    int m = m0 + lr;
    for (int ks = 0; ks < nsteps; ++ks) {
        int cur = ks & 1;
        if (ks + 1 < nsteps && wid == 0) {
            // prefetch next B tile into the other LDS buffer while this one is consumed
            tdm_load_2d_bf16(lds_offset_of(&bufs[cur ^ 1][0]),
                             Bb + (size_t)nblk * K + (ks + 1) * 32,
                             (u32)K, (u32)N, 32u, 256u, (u32)K);
        }
        int kk = ks * 32;
        const bf16* arow = A + (size_t)m * K + kk;
        __builtin_prefetch(arow + 32, 0, 0);
        v16bf a = load16_2x8(arow + 8 * hb, arow + 16 + 8 * hb);
        const bf16* bb = &bufs[cur][0];
#pragma unroll
        for (int t = 0; t < 4; ++t) {
            const v16bf bfrag = load16_2x8(bb + (nl0 + 16 * t + lr) * 32 + 8 * hb, bb + (nl0 + 16 * t + lr) * 32 + 16 + 8 * hb);
            acc[t] = __builtin_amdgcn_wmma_f32_16x16x32_bf16(false, a, false, bfrag,
                                                             (short)0, acc[t], false, false);
        }
        if (ks + 1 < nsteps) {
            if (wid == 0) __builtin_amdgcn_s_wait_tensorcnt(0);
            __syncthreads();
        }
    }

    size_t cb = (size_t)bz * M * N;
#pragma unroll
    for (int t = 0; t < 4; ++t) {
        int n = nblk + nl0 + 16 * t + lr;
#pragma unroll
        for (int r = 0; r < 8; ++r) {
            size_t off = cb + (size_t)(m0 + r + 8 * hb) * N + n;
            if (OUTF32) Cf[off] = acc[t][r];
            else        Cb[off] = (bf16)acc[t][r];
        }
    }
}

// ---------------- repack Q/K into window layout [comp][win][head][tok][dch] ----------------
__global__ void k_repack(const bf16* __restrict__ qkv, bf16* __restrict__ qk) {
    long long idx = (long long)blockIdx.x * blockDim.x + threadIdx.x;
    if (idx >= (long long)2 * NWIN * NHEAD * NTOK * HDIM) return;
    int dch  = (int)(idx & 31);
    int tok  = (int)((idx >> 5) & 63);
    int head = (int)((idx >> 11) & 7);
    int win  = (int)((idx >> 14) & 1023);
    int comp = (int)(idx >> 24);
    int b = win >> 8, wy = (win >> 4) & 15, wx = win & 15;
    int y = wy * 8 + (tok >> 3), x = wx * 8 + (tok & 7);
    int ch = comp * CHN + head * HDIM + dch;        // (comp,h,d) decomposition
    qk[idx] = qkv[((size_t)(b * 768 + ch)) * NPIX + y * 128 + x];
}

// ---------------- pyramid adaptive avg pool (scales 1,2,3,6 over 4 chunks) ----------------
__global__ void k_pool(const bf16* __restrict__ qkv, float* __restrict__ pool) {
    int id = blockIdx.x * blockDim.x + threadIdx.x;
    if (id >= 12800) return;
    int chunk, s, base, loc;
    if (id < 256)       { chunk = 0; s = 1; base = 0;    loc = id; }
    else if (id < 1280) { chunk = 1; s = 2; base = 256;  loc = id - 256; }
    else if (id < 3584) { chunk = 2; s = 3; base = 1280; loc = id - 1280; }
    else                { chunk = 3; s = 6; base = 3584; loc = id - 3584; }
    int cell = loc % (s * s), bc = loc / (s * s);
    int b = bc >> 6, c = bc & 63;
    int i = cell / s, j = cell % s;
    int h0 = (i * 128) / s, h1 = ((i + 1) * 128 + s - 1) / s;
    int w0 = (j * 128) / s, w1 = ((j + 1) * 128 + s - 1) / s;
    int cv = chunk * 64 + c;
    int head = cv >> 5, dch = cv & 31;
    const bf16* src = qkv + ((size_t)(b * 768 + 512 + dch * 8 + head)) * NPIX; // V, (d,h) decomp
    float sum = 0.f;
    for (int yy = h0; yy < h1; ++yy)
        for (int xx = w0; xx < w1; ++xx) sum += (float)src[yy * 128 + xx];
    pool[base + (size_t)bc * s * s + cell] = sum / (float)((h1 - h0) * (w1 - w0));
}

// ---------------- df = df_q * HP + LP, packed TRANSPOSED to [win][head][dch][tok] ----------------
__global__ void k_dfbuild(const bf16* __restrict__ qkv,
                          const float* __restrict__ f3w, const float* __restrict__ f3b,
                          const float* __restrict__ f5w, const float* __restrict__ f5b,
                          const float* __restrict__ f7w, const float* __restrict__ f7b,
                          const float* __restrict__ pool, bf16* __restrict__ dfp) {
    long long idx = (long long)blockIdx.x * blockDim.x + threadIdx.x;
    if (idx >= (long long)BATCH * CHN * NPIX) return;
    int x = (int)(idx & 127), y = (int)((idx >> 7) & 127);
    int cv = (int)((idx >> 14) & 255);
    int b  = (int)(idx >> 22);
    int head = cv >> 5, dch = cv & 31;
    const bf16* vimg = qkv + ((size_t)(b * 768 + 512 + dch * 8 + head)) * NPIX;
    // high-pass: depthwise conv (zero pad)
    int ksz; const float* w; float hp;
    if (cv < 64)       { ksz = 3; w = f3w + cv * 9;          hp = f3b[cv]; }
    else if (cv < 160) { ksz = 5; w = f5w + (cv - 64) * 25;  hp = f5b[cv - 64]; }
    else               { ksz = 7; w = f7w + (cv - 160) * 49; hp = f7b[cv - 160]; }
    int pad = ksz >> 1;
    for (int ky = 0; ky < ksz; ++ky) {
        int iy = y - pad + ky; if ((unsigned)iy >= 128u) continue;
        for (int kx = 0; kx < ksz; ++kx) {
            int ix = x - pad + kx; if ((unsigned)ix >= 128u) continue;
            hp += w[ky * ksz + kx] * (float)vimg[iy * 128 + ix];
        }
    }
    // low-pass: bilinear upsample of pooled pyramid, relu
    const int ssz[4]  = {1, 2, 3, 6};
    const int poff[4] = {0, 256, 1280, 3584};
    int chunk = cv >> 6, s = ssz[chunk];
    const float* P = pool + poff[chunk] + ((size_t)(b * 64 + (cv & 63))) * s * s;
    float ratio = (float)s / 128.0f;
    float fy = (y + 0.5f) * ratio - 0.5f; if (fy < 0.f) fy = 0.f; if (fy > s - 1) fy = (float)(s - 1);
    float fx = (x + 0.5f) * ratio - 0.5f; if (fx < 0.f) fx = 0.f; if (fx > s - 1) fx = (float)(s - 1);
    int y0 = (int)fy, x0 = (int)fx;
    int y1 = y0 + 1; if (y1 > s - 1) y1 = s - 1;
    int x1 = x0 + 1; if (x1 > s - 1) x1 = s - 1;
    float wy = fy - y0, wx = fx - x0;
    float lp = (1 - wy) * ((1 - wx) * P[y0 * s + x0] + wx * P[y0 * s + x1]) +
               wy       * ((1 - wx) * P[y1 * s + x0] + wx * P[y1 * s + x1]);
    lp = fmaxf(lp, 0.f);
    float dfq = (float)qkv[((size_t)(b * 768 + dch * 8 + head)) * NPIX + y * 128 + x];
    float v = dfq * hp + lp;
    int wy_ = y >> 3, wx_ = x >> 3, tok = (y & 7) * 8 + (x & 7);
    int win = (b * 16 + wy_) * 16 + wx_;
    // transposed: [task][dch][tok] so PV B-fragments are tok-contiguous
    dfp[((size_t)(win * 8 + head)) * 2048 + dch * 64 + tok] = (bf16)v;
}

// ---------------- windowed attention: QK^T (wmma) + bias + softmax + PV (wmma) ----------------
__global__ __launch_bounds__(128) void k_attn(const bf16* __restrict__ qk,
                                              const bf16* __restrict__ dfp,
                                              const float* __restrict__ relt,
                                              float* __restrict__ img) {
    __shared__ __align__(32) bf16 pbuf[4][64 * 64];
    int lane = threadIdx.x & 31, wid = threadIdx.x >> 5;
    int task = blockIdx.x * 4 + wid;              // win*8 + head
    int head = task & 7, win = task >> 3;
    int hb = lane >> 4, lr = lane & 15;
    const bf16* Q  = qk + (size_t)task * 2048;
    const bf16* Kp = qk + (size_t)16777216 + (size_t)task * 2048;
    const bf16* V  = dfp + (size_t)task * 2048;   // [dch][tok]

    v8f zero = {};
    v8f acc[4][4];
#pragma unroll
    for (int i = 0; i < 4; ++i)
#pragma unroll
        for (int j = 0; j < 4; ++j) acc[i][j] = zero;

    // S = Q (64x32) * K^T (32x64): B-frag is two 8-elem runs (k=8hb.., 16+8hb..) of a row of K
#pragma unroll
    for (int ti = 0; ti < 4; ++ti) {
        const bf16* qr = Q + (16 * ti + lr) * 32;
        v16bf a = load16_2x8(qr + 8 * hb, qr + 16 + 8 * hb);
#pragma unroll
        for (int tj = 0; tj < 4; ++tj) {
            const bf16* kr = Kp + (16 * tj + lr) * 32;
            v16bf bb = load16_2x8(kr + 8 * hb, kr + 16 + 8 * hb);
            acc[ti][tj] = __builtin_amdgcn_wmma_f32_16x16x32_bf16(false, a, false, bb,
                                                                  (short)0, acc[ti][tj], false, false);
        }
    }

    const float scale = 0.17677669529663687f; // 1/sqrt(32)
#pragma unroll
    for (int ti = 0; ti < 4; ++ti) {
#pragma unroll
        for (int r = 0; r < 8; ++r) {
            int m = 16 * ti + r + 8 * hb;          // query token
            int qy = m >> 3, qx = m & 7;
            float vmax = -3.4e38f;
#pragma unroll
            for (int tj = 0; tj < 4; ++tj) {
                int n = 16 * tj + lr;              // key token
                int ky = n >> 3, kx = n & 7;
                float bias = relt[((qy - ky + 7) * 15 + (qx - kx + 7)) * 8 + head];
                float t = acc[ti][tj][r] * scale + bias;
                acc[ti][tj][r] = t;
                vmax = fmaxf(vmax, t);
            }
            for (int o = 1; o < 16; o <<= 1) vmax = fmaxf(vmax, __shfl_xor(vmax, o, 32));
            float rs = 0.f;
#pragma unroll
            for (int tj = 0; tj < 4; ++tj) {
                float e = __expf(acc[ti][tj][r] - vmax);
                acc[ti][tj][r] = e; rs += e;
            }
            for (int o = 1; o < 16; o <<= 1) rs += __shfl_xor(rs, o, 32);
            float inv = 1.f / rs;
#pragma unroll
            for (int tj = 0; tj < 4; ++tj)
                pbuf[wid][m * 64 + 16 * tj + lr] = (bf16)(acc[ti][tj][r] * inv);
        }
    }

    // O = P (64x64) * V (64x32); V transposed -> contiguous B-frags
    v8f o2[4][2];
#pragma unroll
    for (int i = 0; i < 4; ++i) { o2[i][0] = zero; o2[i][1] = zero; }
#pragma unroll
    for (int kk = 0; kk < 64; kk += 32) {
        v16bf bfrag[2];
#pragma unroll
        for (int tj = 0; tj < 2; ++tj) {
            const bf16* vr = V + (16 * tj + lr) * 64 + kk;
            bfrag[tj] = load16_2x8(vr + 8 * hb, vr + 16 + 8 * hb);
        }
#pragma unroll
        for (int ti = 0; ti < 4; ++ti) {
            const bf16* pr = &pbuf[wid][(16 * ti + lr) * 64 + kk];
            v16bf a = load16_2x8(pr + 8 * hb, pr + 16 + 8 * hb);
#pragma unroll
            for (int tj = 0; tj < 2; ++tj)
                o2[ti][tj] = __builtin_amdgcn_wmma_f32_16x16x32_bf16(false, a, false, bfrag[tj],
                                                                     (short)0, o2[ti][tj], false, false);
        }
    }

    int b = win >> 8, wy = (win >> 4) & 15, wx = win & 15;
#pragma unroll
    for (int ti = 0; ti < 4; ++ti)
#pragma unroll
        for (int tj = 0; tj < 2; ++tj)
#pragma unroll
            for (int r = 0; r < 8; ++r) {
                int m = 16 * ti + r + 8 * hb;
                int dch = 16 * tj + lr;
                int y = wy * 8 + (m >> 3), x = wx * 8 + (m & 7);
                img[(((size_t)(b * CHN + head * HDIM + dch)) * 128 + y) * 128 + x] = o2[ti][tj][r];
            }
}

// ---------------- CBAM: per-channel max/mean over pixels ----------------
__global__ void k_redmm(const bf16* __restrict__ loc1, float* __restrict__ mx, float* __restrict__ av) {
    __shared__ float smax[256], ssum[256];
    int bc = blockIdx.x;                       // b*512 + ch
    int tid = threadIdx.x;
    const bf16* p = loc1 + (size_t)bc * NPIX;
    float m = -3.4e38f, s = 0.f;
    for (int i = tid; i < NPIX; i += 256) { float v = (float)p[i]; m = fmaxf(m, v); s += v; }
    smax[tid] = m; ssum[tid] = s; __syncthreads();
    for (int o = 128; o > 0; o >>= 1) {
        if (tid < o) { smax[tid] = fmaxf(smax[tid], smax[tid + o]); ssum[tid] += ssum[tid + o]; }
        __syncthreads();
    }
    if (tid == 0) { mx[bc] = smax[0]; av[bc] = ssum[0] * (1.0f / NPIX); }
}

__global__ void k_fc(const float* __restrict__ in, const float* __restrict__ w,
                     const float* __restrict__ bias, float* __restrict__ out,
                     int IN, int OUT, int relu) {
    int i = blockIdx.x * blockDim.x + threadIdx.x;
    if (i >= BATCH * OUT) return;
    int b = i / OUT, o = i % OUT;
    float s = bias[o];
    const float* ip = in + (size_t)b * IN;
    const float* wp = w + (size_t)o * IN;
    for (int k = 0; k < IN; ++k) s += ip[k] * wp[k];
    if (relu) s = fmaxf(s, 0.f);
    out[i] = s;
}

__global__ void k_gate(const float* __restrict__ um, const float* __restrict__ ua,
                       float* __restrict__ g) {
    int i = blockIdx.x * blockDim.x + threadIdx.x;
    if (i >= BATCH * 512) return;
    g[i] = sigmoidf_(sigmoidf_(um[i]) + sigmoidf_(ua[i]));
}

__global__ void k_cat(const bf16* __restrict__ loc1, const float* __restrict__ g,
                      float* __restrict__ cat) {
    int n = blockIdx.x * blockDim.x + threadIdx.x;
    if (n >= BATCH * NPIX) return;
    int b = n >> 14, pix = n & 16383;
    const bf16* L = loc1 + (size_t)b * 512 * NPIX + pix;
    const float* gg = g + b * 512;
    float mx = -3.4e38f, sm = 0.f;
    for (int ch = 0; ch < 512; ++ch) {
        float v = gg[ch] * (float)L[(size_t)ch * NPIX];
        mx = fmaxf(mx, v); sm += v;
    }
    cat[((size_t)(b * 2)) * NPIX + pix]     = mx;
    cat[((size_t)(b * 2 + 1)) * NPIX + pix] = sm * (1.0f / 512.0f);
}

__global__ void k_sa(const float* __restrict__ cat, const float* __restrict__ saw,
                     const float* __restrict__ sab, float* __restrict__ sa) {
    int n = blockIdx.x * blockDim.x + threadIdx.x;
    if (n >= BATCH * NPIX) return;
    int b = n >> 14, pix = n & 16383;
    int y = pix >> 7, x = pix & 127;
    float acc = sab[0];
    for (int ci = 0; ci < 2; ++ci)
        for (int ky = 0; ky < 7; ++ky) {
            int iy = y - 3 + ky; if ((unsigned)iy >= 128u) continue;
            for (int kx = 0; kx < 7; ++kx) {
                int ix = x - 3 + kx; if ((unsigned)ix >= 128u) continue;
                acc += saw[(ci * 7 + ky) * 7 + kx] *
                       cat[((size_t)(b * 2 + ci)) * NPIX + iy * 128 + ix];
            }
        }
    sa[(size_t)b * NPIX + pix] = sigmoidf_(acc);
}

// gated loc -> ld2 GEMM input, written transposed [b][pix][512]
__global__ void k_ld2in(const bf16* __restrict__ loc1, const float* __restrict__ g,
                        const float* __restrict__ sa, bf16* __restrict__ out) {
    long long idx = (long long)blockIdx.x * blockDim.x + threadIdx.x;
    if (idx >= (long long)BATCH * 512 * NPIX) return;
    int pix = (int)(idx & 16383);
    int ch  = (int)((idx >> 14) & 511);
    int b   = (int)(idx >> 23);
    float v = (float)loc1[idx] * g[b * 512 + ch] * sa[(size_t)b * NPIX + pix];
    out[((size_t)b * NPIX + pix) * 512 + ch] = (bf16)v;
}

// ---------------- directional avg pooling + local fuse ----------------
__global__ void k_fuse(const float* __restrict__ img, const float* __restrict__ loc,
                       float* __restrict__ out) {
    long long idx = (long long)blockIdx.x * blockDim.x + threadIdx.x;
    if (idx >= (long long)BATCH * CHN * NPIX) return;
    int x = (int)(idx & 127), y = (int)((idx >> 7) & 127);
    const float* I = img + (idx >> 14) * NPIX;   // base of this (b,c) plane
    float s1 = 0.f, s2 = 0.f;
    for (int k = -3; k <= 4; ++k) {
        int t = y + k;
        if (t >= 0 && t <= 128) { int tt = (t == 128) ? 126 : t; s1 += I[tt * 128 + x]; }
    }
    for (int k = -3; k <= 4; ++k) {
        int t = x + k;
        if (t >= 0 && t <= 128) { int tt = (t == 128) ? 126 : t; s2 += I[y * 128 + tt]; }
    }
    out[idx] = (s1 + s2) * 0.125f + loc[idx];
}

// ---------------- reflect-pad(0,1) + depthwise 8x8 (pad 3) + BN -> bf16 NHWC ----------------
__global__ void k_dwbn(const float* __restrict__ fuse, const float* __restrict__ dww,
                       const float* __restrict__ bng, const float* __restrict__ bnb,
                       bf16* __restrict__ out) {
    long long idx = (long long)blockIdx.x * blockDim.x + threadIdx.x;
    if (idx >= (long long)BATCH * CHN * NPIX) return;
    int x = (int)(idx & 127), y = (int)((idx >> 7) & 127);
    int c = (int)((idx >> 14) & 255);
    int b = (int)(idx >> 22);
    const float* F = fuse + (idx >> 14) * NPIX;
    const float* W = dww + c * 64;
    float s = 0.f;
    for (int ky = 0; ky < 8; ++ky) {
        int ty = y - 3 + ky; if (ty < 0 || ty > 128) continue;
        int yy = (ty == 128) ? 126 : ty;
        for (int kx = 0; kx < 8; ++kx) {
            int tx = x - 3 + kx; if (tx < 0 || tx > 128) continue;
            int xx = (tx == 128) ? 126 : tx;
            s += W[ky * 8 + kx] * F[yy * 128 + xx];
        }
    }
    s = s * (bng[c] * rsqrtf(1.0f + 1e-5f)) + bnb[c];
    out[((size_t)b * NPIX + (y * 128 + x)) * 256 + c] = (bf16)s;
}

// ---------------- host launch ----------------
extern "C" void kernel_launch(void* const* d_in, const int* in_sizes, int n_in,
                              void* d_out, int out_size, void* d_ws, size_t ws_size,
                              hipStream_t stream) {
    (void)in_sizes; (void)n_in; (void)out_size; (void)ws_size;
    const float* x        = (const float*)d_in[0];
    const float* qkv_w    = (const float*)d_in[1];
    const float* ld1_w    = (const float*)d_in[2];
    const float* ld2_w    = (const float*)d_in[3];
    const float* camax_w1 = (const float*)d_in[4];
    const float* camax_b1 = (const float*)d_in[5];
    const float* camax_w2 = (const float*)d_in[6];
    const float* camax_b2 = (const float*)d_in[7];
    const float* caavg_w1 = (const float*)d_in[8];
    const float* caavg_b1 = (const float*)d_in[9];
    const float* caavg_w2 = (const float*)d_in[10];
    const float* caavg_b2 = (const float*)d_in[11];
    const float* sa_w     = (const float*)d_in[12];
    const float* sa_b     = (const float*)d_in[13];
    const float* f3_w     = (const float*)d_in[14];
    const float* f3_b     = (const float*)d_in[15];
    const float* f5_w     = (const float*)d_in[16];
    const float* f5_b     = (const float*)d_in[17];
    const float* f7_w     = (const float*)d_in[18];
    const float* f7_b     = (const float*)d_in[19];
    const float* rel_t    = (const float*)d_in[20];
    const float* dw_w     = (const float*)d_in[21];
    const float* bn_g     = (const float*)d_in[22];
    const float* bn_b     = (const float*)d_in[23];
    const float* pw_w     = (const float*)d_in[24];

    char* p = (char*)d_ws;
    auto alloc = [&](size_t bytes) -> void* {
        void* r = (void*)p; p += (bytes + 255) & ~(size_t)255; return r;
    };
    bf16*  wq   = (bf16*)alloc((size_t)768 * 256 * 2);
    bf16*  wl1  = (bf16*)alloc((size_t)512 * 256 * 2);
    bf16*  wl2  = (bf16*)alloc((size_t)256 * 512 * 2);
    bf16*  wpw  = (bf16*)alloc((size_t)256 * 256 * 2);
    bf16*  xb   = (bf16*)alloc((size_t)BATCH * CHN * NPIX * 2);   // NHWC [b][pix][256]
    bf16*  qkvb = (bf16*)alloc((size_t)BATCH * 768 * NPIX * 2);   // NCHW [b][768][pix]
    bf16*  loc1 = (bf16*)alloc((size_t)BATCH * 512 * NPIX * 2);   // NCHW [b][512][pix]
    bf16*  qk   = (bf16*)alloc((size_t)2 * NWIN * NHEAD * 2048 * 2);
    bf16*  dfp  = (bf16*)alloc((size_t)NWIN * NHEAD * 2048 * 2);
    float* pool = (float*)alloc((size_t)12800 * 4);
    float* mx   = (float*)alloc((size_t)2048 * 4);
    float* av   = (float*)alloc((size_t)2048 * 4);
    float* hmax = (float*)alloc((size_t)1024 * 4);
    float* havg = (float*)alloc((size_t)1024 * 4);
    float* umax = (float*)alloc((size_t)2048 * 4);
    float* uavg = (float*)alloc((size_t)2048 * 4);
    float* gbuf = (float*)alloc((size_t)2048 * 4);
    float* cat  = (float*)alloc((size_t)BATCH * 2 * NPIX * 4);
    float* sab  = (float*)alloc((size_t)BATCH * NPIX * 4);
    bf16*  l2in = (bf16*)alloc((size_t)BATCH * 512 * NPIX * 2);   // NHWC [b][pix][512]
    float* loco = (float*)alloc((size_t)BATCH * CHN * NPIX * 4);
    float* img  = (float*)alloc((size_t)BATCH * CHN * NPIX * 4);
    float* fuse = (float*)alloc((size_t)BATCH * CHN * NPIX * 4);
    bf16*  pwin = (bf16*)alloc((size_t)BATCH * CHN * NPIX * 2);   // NHWC [b][pix][256]

    // conversions to bf16 (weights flat, x transposed to NHWC)
    k_f2b<<<(768 * 256 + 255) / 256, 256, 0, stream>>>(qkv_w, wq, 768LL * 256);
    k_f2b<<<(512 * 256 + 255) / 256, 256, 0, stream>>>(ld1_w, wl1, 512LL * 256);
    k_f2b<<<(256 * 512 + 255) / 256, 256, 0, stream>>>(ld2_w, wl2, 256LL * 512);
    k_f2b<<<(256 * 256 + 255) / 256, 256, 0, stream>>>(pw_w, wpw, 256LL * 256);
    k_f2bT<<<(int)(((long long)BATCH * CHN * NPIX + 255) / 256), 256, 0, stream>>>(x, xb, 256);

    // big GEMMs (WMMA + TDM-staged B tiles)
    k_gemm<false><<<dim3(64, 24, BATCH), 256, 0, stream>>>(wq,  xb,   nullptr, qkvb, 768, NPIX, 256);
    k_gemm<false><<<dim3(64, 16, BATCH), 256, 0, stream>>>(wl1, xb,   nullptr, loc1, 512, NPIX, 256);

    // attention branch
    k_repack<<<(int)((2LL * NWIN * NHEAD * 2048) / 256), 256, 0, stream>>>(qkvb, qk);
    k_pool<<<(12800 + 255) / 256, 256, 0, stream>>>(qkvb, pool);
    k_dfbuild<<<(int)(((long long)BATCH * CHN * NPIX) / 256), 256, 0, stream>>>(
        qkvb, f3_w, f3_b, f5_w, f5_b, f7_w, f7_b, pool, dfp);
    k_attn<<<NWIN * NHEAD / 4, 128, 0, stream>>>(qk, dfp, rel_t, img);

    // local (CBAM) branch
    k_redmm<<<BATCH * 512, 256, 0, stream>>>(loc1, mx, av);
    k_fc<<<(BATCH * 256 + 255) / 256, 256, 0, stream>>>(mx, camax_w1, camax_b1, hmax, 512, 256, 1);
    k_fc<<<(BATCH * 512 + 255) / 256, 256, 0, stream>>>(hmax, camax_w2, camax_b2, umax, 256, 512, 0);
    k_fc<<<(BATCH * 256 + 255) / 256, 256, 0, stream>>>(av, caavg_w1, caavg_b1, havg, 512, 256, 1);
    k_fc<<<(BATCH * 512 + 255) / 256, 256, 0, stream>>>(havg, caavg_w2, caavg_b2, uavg, 256, 512, 0);
    k_gate<<<(BATCH * 512 + 255) / 256, 256, 0, stream>>>(umax, uavg, gbuf);
    k_cat<<<(BATCH * NPIX) / 256, 256, 0, stream>>>(loc1, gbuf, cat);
    k_sa<<<(BATCH * NPIX) / 256, 256, 0, stream>>>(cat, sa_w, sa_b, sab);
    k_ld2in<<<(int)(((long long)BATCH * 512 * NPIX) / 256), 256, 0, stream>>>(loc1, gbuf, sab, l2in);
    k_gemm<true><<<dim3(64, 8, BATCH), 256, 0, stream>>>(wl2, l2in, loco, nullptr, 256, NPIX, 512);

    // fuse + projection
    k_fuse<<<(int)(((long long)BATCH * CHN * NPIX) / 256), 256, 0, stream>>>(img, loco, fuse);
    k_dwbn<<<(int)(((long long)BATCH * CHN * NPIX) / 256), 256, 0, stream>>>(fuse, dw_w, bn_g, bn_b, pwin);
    k_gemm<true><<<dim3(64, 8, BATCH), 256, 0, stream>>>(wpw, pwin, (float*)d_out, nullptr, 256, NPIX, 256);
}
